// MAMBACR_3015067042189
// MI455X (gfx1250) — matrix operand order repaired and checked by reading, not compiled, not run on hardware
//
#include <hip/hip_runtime.h>
#include <hip/hip_bf16.h>

// ---------------------------------------------------------------------------
// MambaVision block for MI455X (gfx1250, wave32, WMMA).
// All big GEMMs use v_wmma_f32_16x16x32_f16 (f16 A/B, f32 accum).
// Selective scan: one lane per (n, d_inner, state); 16-lane shfl_xor reduce.
// ---------------------------------------------------------------------------

#define BSZ   8
#define CCH   256
#define LL    4096        // H*W
#define DM    64          // d_model
#define DI    128         // d_inner
#define DST   16          // d_state
#define NBATCH 32         // 4 chunks * 8
#define NT    131072      // NBATCH * LL tokens

typedef __attribute__((ext_vector_type(16))) _Float16 v16h;
typedef __attribute__((ext_vector_type(8)))  float    v8f;

// ---------------- WMMA fragment helpers (CDNA5 ISA layouts) ----------------

// A fragment, 16x32 f16, row-major source with leading dim ld.
// lane m = lane&15 ; lanes 0-15 hold K {0..7,16..23}, lanes 16-31 K {8..15,24..31}
__device__ __forceinline__ v16h fragA(const _Float16* __restrict__ p, int ld) {
  const int lane = threadIdx.x & 31;
  const int m  = lane & 15;
  const int kg = (lane >> 4) * 8;
  v16h f;
#pragma unroll
  for (int i = 0; i < 8; ++i) f[i]     = p[m * ld + kg + i];
#pragma unroll
  for (int i = 0; i < 8; ++i) f[8 + i] = p[m * ld + 16 + kg + i];
  return f;
}

// B fragment, 32x16 f16, built from W stored [N,K] row-major (B = W^T).
// lane n = lane&15 ; lanes 0-15 hold K {0..7,16..23}, lanes 16-31 K {8..15,24..31}
__device__ __forceinline__ v16h fragB(const _Float16* __restrict__ p, int ld) {
  const int lane = threadIdx.x & 31;
  const int n  = lane & 15;
  const int k0 = (lane >> 4) * 8;
  v16h f;
#pragma unroll
  for (int i = 0; i < 8; ++i) f[i] = p[n * ld + k0 + i];
#pragma unroll
  for (int i = 0; i < 8; ++i) f[8 + i] = p[n * ld + 16 + k0 + i];
  return f;
}

__device__ __forceinline__ v8f wmma(v16h a, v16h b, v8f c) {
  return __builtin_amdgcn_wmma_f32_16x16x32_f16(false, a, false, b,
                                                (short)0, c, false, false);
}

__device__ __forceinline__ float fsilu(float x) {
  return x / (1.f + __expf(-x));
}
__device__ __forceinline__ float fgelu(float x) {
  return 0.5f * x * (1.f + erff(x * 0.70710678118f));
}

// ---------------- 0) f32 -> f16 weight conversion ----------------
__global__ void k_cvt(const float* __restrict__ s, _Float16* __restrict__ d, int n) {
  int i = blockIdx.x * 256 + threadIdx.x;
  if (i < n) d[i] = (_Float16)s[i];
}

// ---------------- 1) LayerNorm over C=256 + chunk-split reorder ----------------
// x[B,C,L] -> xi[n = chunk*8+b][l][dm] (f32 and f16), LDS-transposed tile.
__global__ void k_ln1(const float* __restrict__ x, const float* __restrict__ nw,
                      const float* __restrict__ nb, float* __restrict__ xi32,
                      _Float16* __restrict__ xi16) {
  __shared__ float tile[256 * 33];
  __shared__ float red[512];
  __shared__ float mean_s[32], rstd_s[32];
  const int b  = blockIdx.x >> 7;
  const int l0 = (blockIdx.x & 127) << 5;
  const int tid = threadIdx.x;
  const int px = tid & 31, ty = tid >> 5;
  const float* xb = x + (size_t)b * CCH * LL + l0;
#pragma unroll
  for (int it = 0; it < 32; ++it) {
    int c = it * 8 + ty;
    tile[c * 33 + px] = xb[(size_t)c * LL + px];
  }
  __syncthreads();
  float s = 0.f, ss = 0.f;
#pragma unroll
  for (int k = 0; k < 32; ++k) {
    float v = tile[(ty + 8 * k) * 33 + px];
    s += v; ss += v * v;
  }
  red[ty * 32 + px] = s;
  red[256 + ty * 32 + px] = ss;
  __syncthreads();
  if (tid < 32) {
    float S = 0.f, SS = 0.f;
#pragma unroll
    for (int k = 0; k < 8; ++k) { S += red[k * 32 + tid]; SS += red[256 + k * 32 + tid]; }
    float m = S * (1.f / 256.f);
    float var = SS * (1.f / 256.f) - m * m;
    mean_s[tid] = m;
    rstd_s[tid] = rsqrtf(var + 1e-5f);
  }
  __syncthreads();
  const int dm = tid & 63, g = tid >> 6;
#pragma unroll
  for (int it = 0; it < 32; ++it) {
    int chunk = it >> 3;
    int pos = ((it & 7) << 2) | g;
    int c = chunk * 64 + dm;
    float v = (tile[c * 33 + pos] - mean_s[pos]) * rstd_s[pos] * nw[c] + nb[c];
    size_t off = (((size_t)(chunk * 8 + b) * LL) + (l0 + pos)) * 64 + dm;
    xi32[off] = v;
    xi16[off] = (_Float16)v;
  }
}

// ---------------- 2) in_proj GEMM [NT,64]x[64->256] -> xc f32, z f16 ----------------
__global__ void k_inproj(const _Float16* __restrict__ A, const _Float16* __restrict__ W,
                         float* __restrict__ xc, _Float16* __restrict__ z16) {
  const int wave = (blockIdx.x * blockDim.x + threadIdx.x) >> 5;
  const int row0 = wave * 16;
  if (row0 >= NT) return;
  v16h a0 = fragA(A + (size_t)row0 * 64, 64);
  v16h a1 = fragA(A + (size_t)row0 * 64 + 32, 64);
  const int lane = threadIdx.x & 31;
  const int dn = lane & 15, dm0 = (lane >> 4) * 8;
#pragma unroll
  for (int nt = 0; nt < 16; ++nt) {
    v8f acc = {};
    acc = wmma(a0, fragB(W + nt * 16 * 64, 64), acc);
    acc = wmma(a1, fragB(W + nt * 16 * 64 + 32, 64), acc);
    const int col = nt * 16 + dn;
#pragma unroll
    for (int r = 0; r < 8; ++r) {
      const size_t row = row0 + dm0 + r;
      if (col < DI) xc[row * DI + col] = acc[r];
      else          z16[row * DI + (col - DI)] = (_Float16)acc[r];
    }
  }
}

// ---------------- 3) depthwise causal conv(4) + SiLU -> u f16 ----------------
__global__ void k_conv(const float* __restrict__ xc, const float* __restrict__ cw,
                       const float* __restrict__ cb, _Float16* __restrict__ u16) {
  const int tid = blockIdx.x * 256 + threadIdx.x;   // NT*DI threads
  const int d = tid & 127;
  const int token = tid >> 7;
  const int l = token & (LL - 1);
  float acc = cb[d];
#pragma unroll
  for (int t = 0; t < 4; ++t) {
    int ls = l - 3 + t;
    float xv = (ls >= 0) ? xc[(size_t)(token - 3 + t) * DI + d] : 0.f;
    acc += xv * cw[d * 4 + t];
  }
  u16[(size_t)tid] = (_Float16)fsilu(acc);
}

// ---------------- 4) x_proj (128->36) + dt_proj (4->128) + softplus ----------------
__global__ void k_xproj(const _Float16* __restrict__ u16, const float* __restrict__ xw,
                        const float* __restrict__ dtw, const float* __restrict__ dtb,
                        float* __restrict__ dt, float* __restrict__ Bo,
                        float* __restrict__ Co) {
  __shared__ float s_xw[36 * 128];
  __shared__ float s_dtw[128 * 4];
  __shared__ float s_dtb[128];
  for (int i = threadIdx.x; i < 36 * 128; i += 256) s_xw[i] = xw[i];
  for (int i = threadIdx.x; i < 512; i += 256) s_dtw[i] = dtw[i];
  if (threadIdx.x < 128) s_dtb[threadIdx.x] = dtb[threadIdx.x];
  __syncthreads();
  const int token = blockIdx.x * 256 + threadIdx.x;
  const _Float16* up = u16 + (size_t)token * DI;
  float acc[36];
#pragma unroll
  for (int j = 0; j < 36; ++j) acc[j] = 0.f;
  for (int k = 0; k < 128; ++k) {
    float uk = (float)up[k];
#pragma unroll
    for (int j = 0; j < 36; ++j) acc[j] += uk * s_xw[j * 128 + k];
  }
#pragma unroll
  for (int j = 0; j < 16; ++j) Bo[(size_t)token * 16 + j] = acc[4 + j];
#pragma unroll
  for (int j = 0; j < 16; ++j) Co[(size_t)token * 16 + j] = acc[20 + j];
  for (int d = 0; d < 128; ++d) {
    float v = acc[0] * s_dtw[d * 4] + acc[1] * s_dtw[d * 4 + 1] +
              acc[2] * s_dtw[d * 4 + 2] + acc[3] * s_dtw[d * 4 + 3] + s_dtb[d];
    float sp = (v > 20.f) ? v : log1pf(__expf(v));
    dt[(size_t)token * DI + d] = sp;
  }
}

// ---------------- 5) selective scan: lane = (n, d, state) ----------------
__global__ void k_scan(const float* __restrict__ dt, const float* __restrict__ Bm,
                       const float* __restrict__ Cm, const _Float16* __restrict__ u16,
                       const float* __restrict__ A_log, float* __restrict__ y) {
  const int tid = blockIdx.x * 256 + threadIdx.x;   // 65536 threads
  const int s = tid & 15;
  const int d = (tid >> 4) & 127;
  const int n = tid >> 11;
  const float a = -__expf(A_log[d * 16 + s]);
  const float*     dtp = dt  + (size_t)n * LL * DI + d;
  const _Float16*  up  = u16 + (size_t)n * LL * DI + d;
  const float*     bp  = Bm  + (size_t)n * LL * DST + s;
  const float*     cp  = Cm  + (size_t)n * LL * DST + s;
  float h = 0.f;
  float dt_v = dtp[0], u_v = (float)up[0], b_v = bp[0], c_v = cp[0];
  for (int l = 0; l < LL; ++l) {
    const int ln = (l + 1 < LL) ? (l + 1) : (LL - 1);   // software pipeline
    float dtn = dtp[(size_t)ln * DI];
    float un  = (float)up[(size_t)ln * DI];
    float bn  = bp[(size_t)ln * DST];
    float cn  = cp[(size_t)ln * DST];
    h = __expf(dt_v * a) * h + (dt_v * u_v) * b_v;
    float p = h * c_v;
    p += __shfl_xor(p, 8, 16);
    p += __shfl_xor(p, 4, 16);
    p += __shfl_xor(p, 2, 16);
    p += __shfl_xor(p, 1, 16);
    if (s == 0) y[((size_t)n * LL + l) * DI + d] = p;
    dt_v = dtn; u_v = un; b_v = bn; c_v = cn;
  }
}

// ---------------- 6) gate: yg = (y + u*D) * silu(z)  (in place over z16) ----------------
__global__ void k_gate(const float* __restrict__ y, const _Float16* __restrict__ u16,
                       const float* __restrict__ Dp, _Float16* __restrict__ zyg) {
  const int tid = blockIdx.x * 256 + threadIdx.x;   // NT*DI
  const int d = tid & 127;
  float z = (float)zyg[(size_t)tid];
  float yv = y[(size_t)tid] + (float)u16[(size_t)tid] * Dp[d];
  zyg[(size_t)tid] = (_Float16)(yv * fsilu(z));
}

// ---------------- 7) out_proj GEMM [NT,128]x[128->64] -> mo f32 ----------------
__global__ void k_outproj(const _Float16* __restrict__ A, const _Float16* __restrict__ W,
                          float* __restrict__ out) {
  const int wave = (blockIdx.x * blockDim.x + threadIdx.x) >> 5;
  const int row0 = wave * 16;
  if (row0 >= NT) return;
  v16h a[4];
#pragma unroll
  for (int kt = 0; kt < 4; ++kt) a[kt] = fragA(A + (size_t)row0 * 128 + kt * 32, 128);
  const int lane = threadIdx.x & 31;
  const int dn = lane & 15, dm0 = (lane >> 4) * 8;
#pragma unroll
  for (int nt = 0; nt < 4; ++nt) {
    v8f acc = {};
#pragma unroll
    for (int kt = 0; kt < 4; ++kt)
      acc = wmma(a[kt], fragB(W + nt * 16 * 128 + kt * 32, 128), acc);
#pragma unroll
    for (int r = 0; r < 8; ++r)
      out[(size_t)(row0 + dm0 + r) * 64 + nt * 16 + dn] = acc[r];
  }
}

// ---------------- 8) LayerNorm over d_model=64 -> h16 ----------------
__global__ void k_ln2(const float* __restrict__ mo, const float* __restrict__ w,
                      const float* __restrict__ b, _Float16* __restrict__ h16) {
  const int token = blockIdx.x * 256 + threadIdx.x;
  const float* p = mo + (size_t)token * 64;
  float s = 0.f, ss = 0.f;
#pragma unroll
  for (int k = 0; k < 64; ++k) { float v = p[k]; s += v; ss += v * v; }
  float m = s * (1.f / 64.f);
  float rstd = rsqrtf(ss * (1.f / 64.f) - m * m + 1e-5f);
  _Float16* o = h16 + (size_t)token * 64;
#pragma unroll
  for (int k = 0; k < 64; ++k)
    o[k] = (_Float16)((p[k] - m) * rstd * w[k] + b[k]);
}

// ---------------- 9) fc1 GEMM [NT,64]x[64->256] + bias + GELU -> g16 ----------------
__global__ void k_fc1(const _Float16* __restrict__ A, const _Float16* __restrict__ W,
                      const float* __restrict__ bias, _Float16* __restrict__ out) {
  const int wave = (blockIdx.x * blockDim.x + threadIdx.x) >> 5;
  const int row0 = wave * 16;
  if (row0 >= NT) return;
  v16h a0 = fragA(A + (size_t)row0 * 64, 64);
  v16h a1 = fragA(A + (size_t)row0 * 64 + 32, 64);
  const int lane = threadIdx.x & 31;
  const int dn = lane & 15, dm0 = (lane >> 4) * 8;
#pragma unroll
  for (int nt = 0; nt < 16; ++nt) {
    v8f acc = {};
    acc = wmma(a0, fragB(W + nt * 16 * 64, 64), acc);
    acc = wmma(a1, fragB(W + nt * 16 * 64 + 32, 64), acc);
    const int col = nt * 16 + dn;
    const float bb = bias[col];
#pragma unroll
    for (int r = 0; r < 8; ++r)
      out[(size_t)(row0 + dm0 + r) * 256 + col] = (_Float16)fgelu(acc[r] + bb);
  }
}

// ---------------- 10) fc2 GEMM [NT,256]x[256->64] + bias + skip -> out4 f32 ----------------
__global__ void k_fc2(const _Float16* __restrict__ A, const _Float16* __restrict__ W,
                      const float* __restrict__ bias, const float* __restrict__ skip,
                      const float* __restrict__ xi32, float* __restrict__ out) {
  const int wave = (blockIdx.x * blockDim.x + threadIdx.x) >> 5;
  const int row0 = wave * 16;
  if (row0 >= NT) return;
  v16h a[8];
#pragma unroll
  for (int kt = 0; kt < 8; ++kt) a[kt] = fragA(A + (size_t)row0 * 256 + kt * 32, 256);
  const float sk = skip[0];
  const int lane = threadIdx.x & 31;
  const int dn = lane & 15, dm0 = (lane >> 4) * 8;
#pragma unroll
  for (int nt = 0; nt < 4; ++nt) {
    v8f acc = {};
#pragma unroll
    for (int kt = 0; kt < 8; ++kt)
      acc = wmma(a[kt], fragB(W + nt * 16 * 256 + kt * 32, 256), acc);
    const int col = nt * 16 + dn;
    const float bb = bias[col];
#pragma unroll
    for (int r = 0; r < 8; ++r) {
      const size_t row = row0 + dm0 + r;
      out[row * 64 + col] = acc[r] + bb + sk * xi32[row * 64 + col];
    }
  }
}

// ---------------- 11) channel interleave -> inter16 [b*L + hw][c] ----------------
__global__ void k_inter(const float* __restrict__ out4, _Float16* __restrict__ inter16) {
  const int tid = blockIdx.x * 256 + threadIdx.x;   // 8*4096*256
  const int c = tid & 255;
  const int t = tid >> 8;
  const int b = t >> 12;
  const int hw = t & 4095;
  const int chunk = c & 3, dm = c >> 2;
  const int n = chunk * 8 + b;
  inter16[(size_t)tid] = (_Float16)out4[((size_t)n * LL + hw) * 64 + dm];
}

// ---------------- 12) 1x1 conv GEMM [32768,256]x[256->256] + bias -> y1 f32 ----------------
__global__ void k_conv1x1(const _Float16* __restrict__ A, const _Float16* __restrict__ W,
                          const float* __restrict__ bias, float* __restrict__ out) {
  const int wave = (blockIdx.x * blockDim.x + threadIdx.x) >> 5;
  const int row0 = wave * 16;                       // M = 32768
  if (row0 >= BSZ * LL) return;
  v16h a[8];
#pragma unroll
  for (int kt = 0; kt < 8; ++kt) a[kt] = fragA(A + (size_t)row0 * 256 + kt * 32, 256);
  const int lane = threadIdx.x & 31;
  const int dn = lane & 15, dm0 = (lane >> 4) * 8;
#pragma unroll
  for (int nt = 0; nt < 16; ++nt) {
    v8f acc = {};
#pragma unroll
    for (int kt = 0; kt < 8; ++kt)
      acc = wmma(a[kt], fragB(W + nt * 16 * 256 + kt * 32, 256), acc);
    const int col = nt * 16 + dn;
    const float bb = bias[col];
#pragma unroll
    for (int r = 0; r < 8; ++r)
      out[(size_t)(row0 + dm0 + r) * 256 + col] = acc[r] + bb;
  }
}

// ---------------- 13) BN stats (deterministic 2-stage) ----------------
__global__ void k_stats(const float* __restrict__ y1, float* __restrict__ part) {
  const int c = threadIdx.x;
  const int t0 = blockIdx.x * 128;                  // 256 blocks x 128 tokens
  float s = 0.f, ss = 0.f;
  for (int k = 0; k < 128; ++k) {
    float v = y1[(size_t)(t0 + k) * 256 + c];
    s += v; ss += v * v;
  }
  part[blockIdx.x * 256 + c] = s;
  part[65536 + blockIdx.x * 256 + c] = ss;
}
__global__ void k_reduce(const float* __restrict__ part, float* __restrict__ stats) {
  const int c = threadIdx.x;
  float s = 0.f, ss = 0.f;
  for (int k = 0; k < 256; ++k) {
    s  += part[k * 256 + c];
    ss += part[65536 + k * 256 + c];
  }
  stats[c] = s;
  stats[256 + c] = ss;
}

// ---------------- 14) BN apply + ReLU -> d_out [B,C,H,W] ----------------
__global__ void k_bn(const float* __restrict__ y1, const float* __restrict__ stats,
                     const float* __restrict__ gamma, const float* __restrict__ beta,
                     float* __restrict__ out) {
  const int tid = blockIdx.x * 256 + threadIdx.x;   // 8*256*4096
  const int hw = tid & 4095;
  const int c = (tid >> 12) & 255;
  const int b = tid >> 20;
  const float mean = stats[c] * (1.f / 32768.f);
  const float var = stats[256 + c] * (1.f / 32768.f) - mean * mean;
  float v = y1[((size_t)b * 4096 + hw) * 256 + c];
  v = (v - mean) * rsqrtf(var + 1e-5f) * gamma[c] + beta[c];
  out[(size_t)tid] = fmaxf(v, 0.f);
}

// ---------------- launch ----------------
extern "C" void kernel_launch(void* const* d_in, const int* in_sizes, int n_in,
                              void* d_out, int out_size, void* d_ws, size_t ws_size,
                              hipStream_t stream) {
  const float* x         = (const float*)d_in[0];
  const float* norm_w    = (const float*)d_in[1];
  const float* norm_b    = (const float*)d_in[2];
  const float* norm1_w   = (const float*)d_in[3];
  const float* norm1_b   = (const float*)d_in[4];
  const float* in_proj_w = (const float*)d_in[5];
  const float* conv1d_w  = (const float*)d_in[6];
  const float* conv1d_b  = (const float*)d_in[7];
  const float* x_proj_w  = (const float*)d_in[8];
  const float* dt_proj_w = (const float*)d_in[9];
  const float* dt_proj_b = (const float*)d_in[10];
  const float* A_log     = (const float*)d_in[11];
  const float* D_param   = (const float*)d_in[12];
  const float* out_proj_w= (const float*)d_in[13];
  const float* fc1_w     = (const float*)d_in[14];
  const float* fc1_b     = (const float*)d_in[15];
  const float* fc2_w     = (const float*)d_in[16];
  const float* fc2_b     = (const float*)d_in[17];
  const float* skip_sc   = (const float*)d_in[18];
  const float* out_conv_w= (const float*)d_in[19];
  const float* out_conv_b= (const float*)d_in[20];
  const float* bn_gamma  = (const float*)d_in[21];
  const float* bn_beta   = (const float*)d_in[22];
  (void)in_sizes; (void)n_in; (void)out_size; (void)ws_size;

  char* ws = (char*)d_ws;
  // workspace arena (regions reused across pipeline phases)
  float*    xi32   = (float*)   (ws + 0);            // 33.5 MB, live to fc2
  _Float16* xi16   = (_Float16*)(ws + 33554432);     // 16.8 MB, later h16
  float*    xc     = (float*)   (ws + 50331648);     // 67 MB: xc -> dt -> out4
  _Float16* z16    = (_Float16*)(ws + 117440512);    // 33.5 MB: z -> yg -> y1
  _Float16* u16    = (_Float16*)(ws + 150994944);    // 33.5 MB: u -> mo
  float*    Bo     = (float*)   (ws + 184549376);    // 8.4 MB, later inter16
  float*    Co     = (float*)   (ws + 184549376 + 8388608);
  float*    ybuf   = (float*)   (ws + 201326592);    // 67 MB: scan y -> g16
  _Float16* w16    = (_Float16*)(ws + 268435456);    // 240 KB f16 weights
  float*    part   = (float*)   (ws + 268681216);    // 512 KB partials + 2KB stats
  float*    stats  = part + 131072;

  _Float16* w_inproj = w16;
  _Float16* w_outpj  = w16 + 16384;
  _Float16* w_fc1    = w16 + 24576;
  _Float16* w_fc2    = w16 + 40960;
  _Float16* w_cnv    = w16 + 57344;

  // aliases over reused regions
  float*    dtbuf  = xc;                              // after conv consumed xc
  float*    mo     = (float*)u16;                     // after gate consumed u
  _Float16* h16    = xi16;                            // after in_proj consumed xi16
  _Float16* g16    = (_Float16*)ybuf;                 // after gate consumed y
  float*    out4   = xc;                              // after scan consumed dt
  _Float16* inter16= (_Float16*)Bo;                   // after scan consumed B/C
  float*    y1     = (float*)z16;                     // after out_proj consumed yg

  // weights -> f16
  k_cvt<<<64, 256, 0, stream>>>(in_proj_w, w_inproj, 16384);
  k_cvt<<<32, 256, 0, stream>>>(out_proj_w, w_outpj, 8192);
  k_cvt<<<64, 256, 0, stream>>>(fc1_w, w_fc1, 16384);
  k_cvt<<<64, 256, 0, stream>>>(fc2_w, w_fc2, 16384);
  k_cvt<<<256, 256, 0, stream>>>(out_conv_w, w_cnv, 65536);

  k_ln1    <<<1024, 256, 0, stream>>>(x, norm_w, norm_b, xi32, xi16);
  k_inproj <<<1024, 256, 0, stream>>>(xi16, w_inproj, xc, z16);
  k_conv   <<<65536, 256, 0, stream>>>(xc, conv1d_w, conv1d_b, u16);
  k_xproj  <<<512, 256, 0, stream>>>(u16, x_proj_w, dt_proj_w, dt_proj_b,
                                     dtbuf, Bo, Co);
  k_scan   <<<256, 256, 0, stream>>>(dtbuf, Bo, Co, u16, A_log, ybuf);
  k_gate   <<<65536, 256, 0, stream>>>(ybuf, u16, D_param, z16);
  k_outproj<<<1024, 256, 0, stream>>>(z16, w_outpj, mo);
  k_ln2    <<<512, 256, 0, stream>>>(mo, norm1_w, norm1_b, h16);
  k_fc1    <<<1024, 256, 0, stream>>>(h16, w_fc1, fc1_b, g16);
  k_fc2    <<<1024, 256, 0, stream>>>(g16, w_fc2, fc2_b, skip_sc, xi32, out4);
  k_inter  <<<32768, 256, 0, stream>>>(out4, inter16);
  k_conv1x1<<<256, 256, 0, stream>>>(inter16, w_cnv, out_conv_b, y1);
  k_stats  <<<256, 256, 0, stream>>>(y1, part);
  k_reduce <<<1, 256, 0, stream>>>(part, stats);
  k_bn     <<<32768, 256, 0, stream>>>(y1, stats, bn_gamma, bn_beta, (float*)d_out);
}
